// ForceField_70480413328112
// MI455X (gfx1250) — matrix operand order repaired and checked by reading, not compiled, not run on hardware
//
#include <hip/hip_runtime.h>
#include <hip/hip_bf16.h>

typedef __attribute__((ext_vector_type(16))) __bf16 v16bf;
typedef __attribute__((ext_vector_type(8)))  __bf16 v8bf;
typedef __attribute__((ext_vector_type(4)))  __bf16 v4bf;
typedef __attribute__((ext_vector_type(8)))  float  v8f;

#define B_ 8
#define L_ 128
#define R_ 2048
#define E_ 32
#define F_ 64
#define RBF_END_F 12.0f
#define ENERGY_SCALE_F 0.01f

// exp2 with direct v_exp_f32 if available; fallback is mathematically identical.
#if __has_builtin(__builtin_amdgcn_exp2f)
#define EXP2F(x) __builtin_amdgcn_exp2f(x)
#else
#define EXP2F(x) __expf((x) * 0.69314718055994530942f)
#endif

// LDS layout for the staged rec tile (bf16): [e][row][f]
// row stride 72 elements (64 + 8 pad) = 144 B = 36 banks -> conflict-free
// ds_load_b128 across the 16 fragment lanes.
#define ROW_STRIDE 72
#define E_STRIDE   (16 * ROW_STRIDE)   // 1152 elements per e-slice
#define SMEM_ELEMS (E_ * E_STRIDE)     // 36864 bf16 = 73728 B

#define LIG_ELEMS ((size_t)B_ * L_ * E_ * F_)   // 2,097,152

__global__ void ff_zero_kernel(float* out) {
    if (threadIdx.x < B_) out[threadIdx.x] = 0.0f;
}

// One-time pre-pass: lig_feat fp32 -> bf16 in workspace (4 elems/thread).
__global__ __launch_bounds__(256) void ff_cvt_lig_kernel(
    const float* __restrict__ in, __bf16* __restrict__ out) {
    const size_t i = ((size_t)blockIdx.x * 256 + threadIdx.x) * 4;
    const float4 f = *(const float4*)(in + i);
    v4bf o;
    o[0] = (__bf16)f.x; o[1] = (__bf16)f.y;
    o[2] = (__bf16)f.z; o[3] = (__bf16)f.w;
    *(v4bf*)(out + i) = o;
}

// Grid: B_ * (R_/16) workgroups of 256 threads (8 waves).
// Wave w handles l-tile [16w, 16w+16); workgroup handles one 16-wide r-tile.
template <bool LIGBF>
__global__ __launch_bounds__(256) void ff_energy_kernel(
    const float*  __restrict__ lig_feat_f32,  // [B, L, E, F] (used if !LIGBF)
    const __bf16* __restrict__ lig_feat_bf,   // [B, L, E, F] (used if LIGBF)
    const float*  __restrict__ rec_feat,      // [B, R, E, F]
    const float*  __restrict__ lig_coord,     // [B, L, 3]
    const float*  __restrict__ rec_coord,     // [B, R, 3]
    float* __restrict__ out)                  // [B]
{
    __shared__ __bf16 smem[SMEM_ELEMS];

    const int wave = threadIdx.x >> 5;          // 0..7 -> l-tile
    const int lane = threadIdx.x & 31;
    const int b    = blockIdx.x / (R_ / 16);
    const int r0   = (blockIdx.x % (R_ / 16)) * 16;
    const int l0   = wave * 16;

    const int half = lane >> 4;                 // lane half selects K-run
    const int lo   = lane & 15;                 // A: M row / B: N col / C: N col

    // ---- cooperative stage: rec tile [16 rows x E x F] fp32 -> bf16 in LDS.
    for (int c = threadIdx.x; c < 16 * E_; c += 256) {
        const int row = c >> 5;                 // 0..15
        const int e   = c & 31;                 // 0..31
        const float4* src = (const float4*)(rec_feat +
            (((size_t)b * R_ + (r0 + row)) * E_ + e) * F_);
        __bf16* dst = smem + e * E_STRIDE + row * ROW_STRIDE;
#pragma unroll
        for (int i = 0; i < 16; ++i) {
            const float4 f = src[i];
            v4bf o;
            o[0] = (__bf16)f.x; o[1] = (__bf16)f.y;
            o[2] = (__bf16)f.z; o[3] = (__bf16)f.w;
            *(v4bf*)(dst + i * 4) = o;
        }
    }

    // Pre-scaled constants: exp(-((d-mu)/sigma)^2) = exp2(-((d-mu)*cs)^2)
    // with cs = |1/sigma| * sqrt(log2(e)); sigma = -12/32.
    const float cs       = (32.0f / 12.0f) * 1.2011224087864498f;
    const float wmu_step = (RBF_END_F / (float)(E_ - 1)) * cs;

    // ---- pre-scaled distances, matching C layout: lane holds N=lo, M=half*8+j
    float wd[8];
    {
        const float* rc = rec_coord + ((size_t)b * R_ + (r0 + lo)) * 3;
        const float rx = rc[0], ry = rc[1], rz = rc[2];
#pragma unroll
        for (int j = 0; j < 8; ++j) {
            const int m = half * 8 + j;
            const float* lc = lig_coord + ((size_t)b * L_ + (l0 + m)) * 3;
            const float dx = lc[0] - rx, dy = lc[1] - ry, dz = lc[2] - rz;
            wd[j] = sqrtf(dx * dx + dy * dy + dz * dz) * cs;
        }
    }

    __syncthreads();   // rec tile staged (single barrier for the whole kernel)

    // A row base for this lane (lanes lo and lo+16 read row M=lo, different K runs)
    const size_t arow_off = (((size_t)b * L_ + (l0 + lo)) * E_) * F_;

    float energy = 0.0f;

    for (int e = 0; e < E_; ++e) {
        const __bf16* brow = smem + e * E_STRIDE + lo * ROW_STRIDE;

        v8f c = {};
#pragma unroll
        for (int kc = 0; kc < 2; ++kc) {        // F=64 = 2 x (K=32)
            const int k0 = kc * 32;
            // A 16x32 bf16 layout: lanes 0-15 hold K=0..7 & 16..23;
            // lanes 16-31 hold K=8..15 & 24..31, same M rows.
            v16bf a;
            if constexpr (LIGBF) {
                const __bf16* apb = lig_feat_bf + arow_off + (size_t)e * F_
                                    + k0 + half * 8;
                const v8bf a0 = *(const v8bf*)(apb);
                const v8bf a1 = *(const v8bf*)(apb + 16);
                a = __builtin_shufflevector(a0, a1,
                    0, 1, 2, 3, 4, 5, 6, 7, 8, 9, 10, 11, 12, 13, 14, 15);
            } else {
                const float* ap = lig_feat_f32 + arow_off + (size_t)e * F_;
#pragma unroll
                for (int i = 0; i < 8; ++i) {
                    a[i]     = (__bf16)ap[k0 + half * 8 + i];
                    a[8 + i] = (__bf16)ap[k0 + 16 + half * 8 + i];
                }
            }
            // B 32x16 bf16 layout: lanes 0-15 hold K=0..7 & 16..23 of column N=lo,
            // lanes 16-31 hold K=8..15 & 24..31 -- pre-converted rows in LDS.
            const __bf16* bp = brow + k0 + half * 8;
            const v8bf b0 = *(const v8bf*)(bp);
            const v8bf b1 = *(const v8bf*)(bp + 16);
            const v16bf bm = __builtin_shufflevector(b0, b1,
                0, 1, 2, 3, 4, 5, 6, 7, 8, 9, 10, 11, 12, 13, 14, 15);
            c = __builtin_amdgcn_wmma_f32_16x16x32_bf16(
                    false, a, false, bm, (short)0, c, false, false);
        }

        const float wmu = wmu_step * (float)e;
#pragma unroll
        for (int j = 0; j < 8; ++j) {
            const float t = wd[j] - wmu;
            energy = fmaf(c[j], EXP2F(-(t * t)), energy);
        }
    }

    energy *= ENERGY_SCALE_F;

    // wave32 reduction, then one atomic per wave
#pragma unroll
    for (int off = 16; off > 0; off >>= 1)
        energy += __shfl_down(energy, off, 32);
    if (lane == 0)
        atomicAdd(&out[b], energy);
}

extern "C" void kernel_launch(void* const* d_in, const int* in_sizes, int n_in,
                              void* d_out, int out_size, void* d_ws, size_t ws_size,
                              hipStream_t stream) {
    (void)in_sizes; (void)n_in; (void)out_size;
    const float* lig_feat  = (const float*)d_in[0];
    const float* rec_feat  = (const float*)d_in[1];
    const float* lig_coord = (const float*)d_in[2];
    const float* rec_coord = (const float*)d_in[3];
    float* out = (float*)d_out;

    ff_zero_kernel<<<1, 32, 0, stream>>>(out);

    if (ws_size >= LIG_ELEMS * sizeof(__bf16) && d_ws != nullptr) {
        __bf16* lig_bf = (__bf16*)d_ws;
        ff_cvt_lig_kernel<<<(int)(LIG_ELEMS / 4 / 256), 256, 0, stream>>>(
            lig_feat, lig_bf);
        ff_energy_kernel<true><<<B_ * (R_ / 16), 256, 0, stream>>>(
            lig_feat, lig_bf, rec_feat, lig_coord, rec_coord, out);
    } else {
        ff_energy_kernel<false><<<B_ * (R_ / 16), 256, 0, stream>>>(
            lig_feat, nullptr, rec_feat, lig_coord, rec_coord, out);
    }
}
